// PCA_GM_Net_28913719836837
// MI455X (gfx1250) — compile-verified, matrix operand order checked by reading, not run on hardware
//
#include <hip/hip_runtime.h>

// ---------------------------------------------------------------------------
// Types for CDNA5 WMMA (wave32, 16x16x32 bf16 -> f32) and TDM descriptors
// ---------------------------------------------------------------------------
typedef __bf16 bhalf;
typedef __attribute__((ext_vector_type(16))) __bf16 v16bf;
typedef __attribute__((ext_vector_type(8)))  __bf16 v8bf;
typedef __attribute__((ext_vector_type(8)))  float  v8f;
typedef __attribute__((ext_vector_type(4)))  unsigned int u32x4;
typedef __attribute__((ext_vector_type(8)))  int i32x8;
typedef __attribute__((ext_vector_type(4)))  int i32x4;

union Frag { v16bf v; v8bf h[2]; };

#if __has_builtin(__builtin_amdgcn_tensor_load_to_lds) && __has_builtin(__builtin_amdgcn_s_wait_tensorcnt)
#define USE_TDM 1
#else
#define USE_TDM 0
#endif

// LDS tile row stride: 32 data halves + 8 pad halves = 80B (matches TDM pad:
// interval 16 DWORDs = 64B, amount 4 DWORDs = 16B). Avoids ds bank aliasing.
#define LDSROW 40

// Generic LDS pointer -> LDS byte offset (ISA 10.2: offset is the low 32 address bits).
static __device__ inline unsigned lds_off(const void* p) {
    return (unsigned)(unsigned long long)(size_t)p;
}

// Issue one TDM 2D tile load: 32 halves x 128 rows, row stride = strideElems,
// hardware-padded into LDS rows of 80B. Wave-level instruction (EXEC ignored).
static __device__ inline void tdm_issue(unsigned ldsByte, const bhalf* gsrc, int strideElems) {
#if USE_TDM
    unsigned long long ga = (unsigned long long)(size_t)gsrc;
    u32x4 g0;
    g0.x = 1u;                                         // count=1, user desc
    g0.y = ldsByte;                                    // lds_addr
    g0.z = (unsigned)ga;                               // global_addr[31:0]
    g0.w = (unsigned)((ga >> 32) & 0x01FFFFFFu) | (2u << 30);  // addr[56:32] | type=2
    const unsigned t0 = 0x40000000u, t1 = 0x40000000u; // huge tensor dims (no OOB clamp)
    i32x8 g1;
    g1[0] = (int)((1u << 16) |                         // data_size = 2B
                  (1u << 20) |                         // pad_enable
                  (3u << 22) |                         // pad_interval: 16 DWORDs (64B)
                  (3u << 25));                         // pad_amount: 4 DWORDs (16B)
    g1[1] = (int)((t0 & 0xFFFFu) << 16);               // tensor_dim0[15:0]
    g1[2] = (int)(((t0 >> 16) & 0xFFFFu) | ((t1 & 0xFFFFu) << 16));
    g1[3] = (int)(((t1 >> 16) & 0xFFFFu) | (32u << 16)); // tile_dim0 = 32
    g1[4] = (int)128u;                                 // tile_dim1 = 128, tile_dim2 = 0
    g1[5] = (int)(unsigned)strideElems;                // tensor_dim0_stride lo
    g1[6] = 0;                                         // stride hi / dim1_stride lo
    g1[7] = 0;
    i32x4 gz4 = {0, 0, 0, 0};
    i32x8 gz8 = {0, 0, 0, 0, 0, 0, 0, 0};
    // 6-arg form (this toolchain): (g0, g1, g2, g3, g4, cpol)
    __builtin_amdgcn_tensor_load_to_lds(g0, g1, gz4, gz4, gz8, 0);
#else
    (void)ldsByte; (void)gsrc; (void)strideElems;
#endif
}

// Stage one 128x32 bf16 tile into LDS (padded rows of LDSROW halves).
static __device__ inline void stage_tile(bhalf* dst, const bhalf* src, int strideElems, int tid) {
#if USE_TDM
    if (tid < 32) tdm_issue(lds_off(dst), src, strideElems);   // wave 0 issues the DMA
#else
    for (int c = tid; c < 512; c += 256) {                      // 512 x 16B chunks
        int r = c >> 2, q = (c & 3) * 8;
        *(v8bf*)(dst + r * LDSROW + q) = *(const v8bf*)(src + (size_t)r * strideElems + q);
    }
#endif
}

static __device__ inline void stage_wait() {
#if USE_TDM
    __builtin_amdgcn_s_wait_tensorcnt(0);
#endif
    __syncthreads();
}

// GEMM epilogue flags
#define GF_RELU 1
#define GF_BIAS 2
#define GF_ADDB 4   // add bf16 C-input [lrow, col]
#define GF_OUTB 8   // store bf16, row-major
#define GF_OUTF 16  // store f32, row-major
#define GF_OUTT 32  // store bf16, per-batch transposed [b][col][n], n = lrow % 256
#define GF_ADDF 64  // add f32 C-input [lrow, col]

// ---------------------------------------------------------------------------
// WMMA GEMM with TDM/LDS double-buffered staging.
//   C[b] = epilogue( A[b] (rows x K, bf16 row-major) x B[b] given as Bt (MxK) )
// Block: 256 threads = 8 waves (2x4), tile 128x128, wave tile 64x32, KT=32.
// ---------------------------------------------------------------------------
template<int FLAGS>
__global__ __launch_bounds__(256)
void gemm_bf16_k(const bhalf* __restrict__ A, long long aBS, int lda,
                 const bhalf* __restrict__ Bt, long long bBS, int ldb,
                 const float* __restrict__ bias,
                 const void* __restrict__ Cin, int ldc,
                 bhalf* __restrict__ outB, float* __restrict__ outF,
                 bhalf* __restrict__ outT, int ldo, int Mtot, int K)
{
    __shared__ __align__(16) bhalf sA[2][128 * LDSROW];
    __shared__ __align__(16) bhalf sB[2][128 * LDSROW];

    const int tid    = threadIdx.x;
    const int lane   = tid & 31;
    const int wid    = tid >> 5;
    const int wm     = wid >> 2;          // 0..1
    const int wn     = wid & 3;           // 0..3
    const int laneRow = lane & 15;
    const int laneK8  = (lane >> 4) << 3; // 0 or 8

    const bhalf* Ab  = A  + (long long)blockIdx.z * aBS + (size_t)(blockIdx.y * 128) * lda;
    const bhalf* Btb = Bt + (long long)blockIdx.z * bBS + (size_t)(blockIdx.x * 128) * ldb;

    v8f zero = {0.f, 0.f, 0.f, 0.f, 0.f, 0.f, 0.f, 0.f};
    v8f acc[4][2];
#pragma unroll
    for (int i = 0; i < 4; ++i)
#pragma unroll
        for (int j = 0; j < 2; ++j) acc[i][j] = zero;

    stage_tile(sA[0], Ab,  lda, tid);
    stage_tile(sB[0], Btb, ldb, tid);

    const int nk = K >> 5;
    for (int ki = 0; ki < nk; ++ki) {
        const int cur = ki & 1;
        stage_wait();                                  // DMA done + tile published
        if (ki + 1 < nk) {
            stage_tile(sA[cur ^ 1], Ab  + ((ki + 1) << 5), lda, tid);
            stage_tile(sB[cur ^ 1], Btb + ((ki + 1) << 5), ldb, tid);
            __builtin_prefetch(Ab + ((ki + 1) << 5) + (size_t)(tid >> 1) * lda, 0, 1);
        }

        Frag a[4], b[2];
#pragma unroll
        for (int sm = 0; sm < 4; ++sm) {
            const bhalf* p = sA[cur] + (wm * 64 + sm * 16 + laneRow) * LDSROW + laneK8;
            a[sm].h[0] = *(const v8bf*)p;
            a[sm].h[1] = *(const v8bf*)(p + 16);
        }
#pragma unroll
        for (int sn = 0; sn < 2; ++sn) {
            const bhalf* p = sB[cur] + (wn * 32 + sn * 16 + laneRow) * LDSROW + laneK8;
            b[sn].h[0] = *(const v8bf*)p;
            b[sn].h[1] = *(const v8bf*)(p + 16);
        }
#pragma unroll
        for (int sm = 0; sm < 4; ++sm)
#pragma unroll
            for (int sn = 0; sn < 2; ++sn)
                acc[sm][sn] = __builtin_amdgcn_wmma_f32_16x16x32_bf16(
                    false, a[sm].v, false, b[sn].v, (short)0, acc[sm][sn], false, false);

        __syncthreads();                               // done reading cur
    }

    // Epilogue. C fragment layout: VGPR r, lanes 0-15 -> M=r, N=lane;
    // lanes 16-31 -> M=r+8, N=lane-16.
    const int rowBase = blockIdx.y * 128 + wm * 64;
    const int colBase = blockIdx.x * 128 + wn * 32;
    const int rOffHi  = (lane >= 16) ? 8 : 0;
#pragma unroll
    for (int sm = 0; sm < 4; ++sm) {
#pragma unroll
        for (int sn = 0; sn < 2; ++sn) {
#pragma unroll
            for (int r = 0; r < 8; ++r) {
                int row = rowBase + sm * 16 + r + rOffHi;
                int col = colBase + sn * 16 + laneRow;
                long long lrow = (long long)blockIdx.z * 256 + row;
                float v = acc[sm][sn][r];
                if (FLAGS & GF_BIAS) v += bias[col];
                if (FLAGS & GF_ADDB) v += (float)((const bhalf*)Cin)[lrow * ldc + col];
                if (FLAGS & GF_ADDF) v += ((const float*)Cin)[lrow * ldc + col];
                if (FLAGS & GF_RELU) v = v > 0.f ? v : 0.f;
                if (FLAGS & GF_OUTB) outB[lrow * ldo + col] = (bhalf)v;
                if (FLAGS & GF_OUTF) outF[lrow * ldo + col] = v;
                if (FLAGS & GF_OUTT)
                    outT[(lrow >> 8) * (long long)Mtot * 256 + (long long)col * 256 + (lrow & 255)] = (bhalf)v;
            }
        }
    }
}

// ---------------------------------------------------------------------------
// Small helper kernels
// ---------------------------------------------------------------------------
__global__ void cast_bf16_k(const float* __restrict__ in, bhalf* __restrict__ out, long long n) {
    long long i = (long long)blockIdx.x * blockDim.x + threadIdx.x;
    if (i < n) out[i] = (bhalf)in[i];
}

// out[m*K + k] = in[(rowOff+k)*M + m]  : f32 [K x M] (row slice) -> bf16 [M x K]
__global__ void tcast_k(const float* __restrict__ in, bhalf* __restrict__ out,
                        int K, int M, int rowOff) {
    long long i = (long long)blockIdx.x * blockDim.x + threadIdx.x;
    if (i >= (long long)K * M) return;
    int m = (int)(i / K);
    int k = (int)(i % K);
    out[i] = (bhalf)in[(long long)(rowOff + k) * M + m];
}

// Column-normalize A (L1 over rows) and cast to bf16. One thread per (b, j).
__global__ void colnorm_k(const float* __restrict__ A, bhalf* __restrict__ An) {
    int idx = blockIdx.x * blockDim.x + threadIdx.x; // 64*256
    int b = idx >> 8, j = idx & 255;
    const float* a = A + (size_t)b * 65536 + j;
    float s = 0.f;
    for (int i = 0; i < 256; ++i) s += fabsf(a[i * 256]);
    float inv = 1.f / fmaxf(s, 1e-12f);
    bhalf* o = An + (size_t)b * 65536 + j;
    for (int i = 0; i < 256; ++i) o[i * 256] = (bhalf)(a[i * 256] * inv);
}

// ---------------------------------------------------------------------------
// Sinkhorn (log-domain, masked). s is [64,256,256] f32, updated in place.
// ---------------------------------------------------------------------------
#define SK_NEG (-1e30f)

__global__ void sink_init_k(float* __restrict__ s, const int* __restrict__ n1,
                            const int* __restrict__ n2, const int* __restrict__ tau_p) {
    int idx = blockIdx.x * 256 + threadIdx.x;
    int b = idx >> 16, i = (idx >> 8) & 255, j = idx & 255;
    float tau = (float)(*tau_p);
    float v = s[idx];
    bool m = (i < n1[b]) && (j < n2[b]);
    s[idx] = m ? v / tau : SK_NEG;
}

__global__ void sink_row_k(float* __restrict__ s, const int* __restrict__ n1) {
    int b = blockIdx.y, i = blockIdx.x, j = threadIdx.x;
    __shared__ float red[256];
    int base = b * 65536 + i * 256;
    float v = s[base + j];
    red[j] = v; __syncthreads();
    for (int o = 128; o; o >>= 1) { if (j < o) red[j] = fmaxf(red[j], red[j + o]); __syncthreads(); }
    float mx = red[0]; __syncthreads();
    red[j] = __expf(v - mx); __syncthreads();
    for (int o = 128; o; o >>= 1) { if (j < o) red[j] += red[j + o]; __syncthreads(); }
    float lse = mx + __logf(red[0]);
    if (i < n1[b]) s[base + j] = v - lse;
}

__global__ void sink_col_k(float* __restrict__ s, const int* __restrict__ n2) {
    int b = blockIdx.y, j = blockIdx.x, i = threadIdx.x;
    __shared__ float red[256];
    int idx = b * 65536 + i * 256 + j;
    float v = s[idx];
    red[i] = v; __syncthreads();
    for (int o = 128; o; o >>= 1) { if (i < o) red[i] = fmaxf(red[i], red[i + o]); __syncthreads(); }
    float mx = red[0]; __syncthreads();
    red[i] = __expf(v - mx); __syncthreads();
    for (int o = 128; o; o >>= 1) { if (i < o) red[i] += red[i + o]; __syncthreads(); }
    float lse = mx + __logf(red[0]);
    if (j < n2[b]) s[idx] = v - lse;
}

__global__ void sink_final_k(const float* __restrict__ s, float* __restrict__ outF,
                             bhalf* __restrict__ outB, bhalf* __restrict__ outT,
                             const int* __restrict__ n1, const int* __restrict__ n2) {
    int idx = blockIdx.x * 256 + threadIdx.x;
    int b = idx >> 16, i = (idx >> 8) & 255, j = idx & 255;
    bool m = (i < n1[b]) && (j < n2[b]);
    float v = m ? __expf(s[idx]) : 0.f;
    if (outF) outF[idx] = v;
    if (outB) outB[idx] = (bhalf)v;
    if (outT) outT[b * 65536 + j * 256 + i] = (bhalf)v;
}

// ---------------------------------------------------------------------------
// Host-side launcher
// ---------------------------------------------------------------------------
template<int F>
static void G(hipStream_t st, dim3 grid,
              const bhalf* A, long long aBS, int lda,
              const bhalf* Bt, long long bBS, int ldb,
              const float* bias, const void* Cin, int ldc,
              bhalf* outB, float* outF, bhalf* outT, int ldo, int Mtot, int K) {
    gemm_bf16_k<F><<<grid, dim3(256), 0, st>>>(A, aBS, lda, Bt, bBS, ldb, bias, Cin, ldc,
                                               outB, outF, outT, ldo, Mtot, K);
}

extern "C" void kernel_launch(void* const* d_in, const int* in_sizes, int n_in,
                              void* d_out, int out_size, void* d_ws, size_t ws_size,
                              hipStream_t stream) {
    (void)in_sizes; (void)n_in; (void)out_size; (void)ws_size;
    const float* feat1   = (const float*)d_in[0];
    const float* feat2   = (const float*)d_in[1];
    const float* A1      = (const float*)d_in[2];
    const float* A2      = (const float*)d_in[3];
    const float* g0_aW   = (const float*)d_in[4];
    const float* g0_ab   = (const float*)d_in[5];
    const float* g0_uW   = (const float*)d_in[6];
    const float* g0_ub   = (const float*)d_in[7];
    const float* g1_aW   = (const float*)d_in[8];
    const float* g1_ab   = (const float*)d_in[9];
    const float* g1_uW   = (const float*)d_in[10];
    const float* g1_ub   = (const float*)d_in[11];
    const float* aff0_W  = (const float*)d_in[12];
    const float* aff1_W  = (const float*)d_in[13];
    const float* cross_W = (const float*)d_in[14];
    const float* cross_b = (const float*)d_in[15];
    const int*   n1      = (const int*)d_in[16];
    const int*   n2      = (const int*)d_in[17];
    const int*   tau_p   = (const int*)d_in[20];
    float* out = (float*)d_out;

    const long long BN = 64LL * 256;       // 16384 activation rows
    const int IN = 1024, HID = 2048;

    // ---- workspace bump allocator (256B aligned) ----
    char* base = (char*)d_ws; size_t off = 0;
    auto alloc = [&](size_t bytes) -> void* {
        off = (off + 255) & ~(size_t)255;
        void* p = base + off; off += bytes; return p;
    };
    bhalf* feat1b = (bhalf*)alloc(BN * IN * 2);
    bhalf* feat2b = (bhalf*)alloc(BN * IN * 2);
    bhalf* w_g0a  = (bhalf*)alloc((size_t)HID * IN * 2);   // [M=2048 x K=1024]
    bhalf* w_g0u  = (bhalf*)alloc((size_t)HID * IN * 2);
    bhalf* w_g1a  = (bhalf*)alloc((size_t)HID * HID * 2);
    bhalf* w_g1u  = (bhalf*)alloc((size_t)HID * HID * 2);
    bhalf* w_af0  = (bhalf*)alloc((size_t)HID * HID * 2);
    bhalf* w_af1  = (bhalf*)alloc((size_t)HID * HID * 2);
    bhalf* w_crT  = (bhalf*)alloc((size_t)HID * HID * 2);
    bhalf* w_crB  = (bhalf*)alloc((size_t)HID * HID * 2);
    bhalf* An1    = (bhalf*)alloc(64LL * 65536 * 2);
    bhalf* An2    = (bhalf*)alloc(64LL * 65536 * 2);
    bhalf* ax1t   = (bhalf*)alloc(BN * HID * 2);           // [64][2048 x 256]
    bhalf* ax2t   = (bhalf*)alloc(BN * HID * 2);
    bhalf* ux1    = (bhalf*)alloc(BN * HID * 2);
    bhalf* ux2    = (bhalf*)alloc(BN * HID * 2);
    bhalf* emb1   = (bhalf*)alloc(BN * HID * 2);
    bhalf* emb1t  = (bhalf*)alloc(BN * HID * 2);
    bhalf* emb2   = (bhalf*)alloc(BN * HID * 2);
    bhalf* emb2t  = (bhalf*)alloc(BN * HID * 2);
    float* sbuf   = (float*)alloc(64LL * 65536 * 4);
    bhalf* s_bf   = (bhalf*)alloc(64LL * 65536 * 2);
    bhalf* s_tbf  = (bhalf*)alloc(64LL * 65536 * 2);
    bhalf* tmp1   = (bhalf*)alloc(BN * HID * 2);
    bhalf* tmp2   = (bhalf*)alloc(BN * HID * 2);
    float* P      = (float*)alloc(BN * HID * 4);
    bhalf* ne1    = (bhalf*)alloc(BN * HID * 2);
    bhalf* ne2    = (bhalf*)alloc(BN * HID * 2);
    bhalf* Tbuf   = ax1t; // reuse: free between emb-GEMM and layer-1 ax GEMM

    dim3 thr(256);
    // ---- precision conversion / weight transposes ----
    cast_bf16_k<<<dim3((unsigned)((BN * IN) / 256)), thr, 0, stream>>>(feat1, feat1b, BN * IN);
    cast_bf16_k<<<dim3((unsigned)((BN * IN) / 256)), thr, 0, stream>>>(feat2, feat2b, BN * IN);
    tcast_k<<<dim3((IN * HID) / 256), thr, 0, stream>>>(g0_aW,  w_g0a, IN,  HID, 0);
    tcast_k<<<dim3((IN * HID) / 256), thr, 0, stream>>>(g0_uW,  w_g0u, IN,  HID, 0);
    tcast_k<<<dim3((HID * HID) / 256), thr, 0, stream>>>(g1_aW,  w_g1a, HID, HID, 0);
    tcast_k<<<dim3((HID * HID) / 256), thr, 0, stream>>>(g1_uW,  w_g1u, HID, HID, 0);
    tcast_k<<<dim3((HID * HID) / 256), thr, 0, stream>>>(aff0_W, w_af0, HID, HID, 0);
    tcast_k<<<dim3((HID * HID) / 256), thr, 0, stream>>>(aff1_W, w_af1, HID, HID, 0);
    tcast_k<<<dim3((HID * HID) / 256), thr, 0, stream>>>(cross_W, w_crT, HID, HID, 0);
    tcast_k<<<dim3((HID * HID) / 256), thr, 0, stream>>>(cross_W, w_crB, HID, HID, HID);
    colnorm_k<<<dim3(64), thr, 0, stream>>>(A1, An1);
    colnorm_k<<<dim3(64), thr, 0, stream>>>(A2, An2);

    dim3 gFlat(HID / 128, (unsigned)(BN / 128), 1);   // 16 x 128
    dim3 gBat (HID / 128, 2, 64);                     // per-batch, 256 rows
    dim3 gS   (256 / 128, 2, 64);

    // ---- layer 0: Gconv ----
    G<GF_RELU | GF_BIAS | GF_OUTT>(stream, gFlat, feat1b, 0, IN, w_g0a, 0, IN,
        g0_ab, nullptr, 0, nullptr, nullptr, ax1t, HID, HID, IN);
    G<GF_RELU | GF_BIAS | GF_OUTB>(stream, gFlat, feat1b, 0, IN, w_g0u, 0, IN,
        g0_ub, nullptr, 0, ux1, nullptr, nullptr, HID, HID, IN);
    G<GF_RELU | GF_BIAS | GF_OUTT>(stream, gFlat, feat2b, 0, IN, w_g0a, 0, IN,
        g0_ab, nullptr, 0, nullptr, nullptr, ax2t, HID, HID, IN);
    G<GF_RELU | GF_BIAS | GF_OUTB>(stream, gFlat, feat2b, 0, IN, w_g0u, 0, IN,
        g0_ub, nullptr, 0, ux2, nullptr, nullptr, HID, HID, IN);
    // emb = An @ ax + ux   (batched; Bt = ax^T per batch)
    G<GF_ADDB | GF_OUTB | GF_OUTT>(stream, gBat, An1, 65536, 256, ax1t, (long long)HID * 256, 256,
        nullptr, ux1, HID, emb1, nullptr, emb1t, HID, HID, 256);
    G<GF_ADDB | GF_OUTB | GF_OUTT>(stream, gBat, An2, 65536, 256, ax2t, (long long)HID * 256, 256,
        nullptr, ux2, HID, emb2, nullptr, emb2t, HID, HID, 256);

    // ---- affinity 0 + Sinkhorn ----
    G<GF_OUTB>(stream, gFlat, emb1, 0, HID, w_af0, 0, HID,
        nullptr, nullptr, 0, Tbuf, nullptr, nullptr, HID, HID, HID);
    G<GF_OUTF>(stream, gS, Tbuf, 256LL * HID, HID, emb2, 256LL * HID, HID,
        nullptr, nullptr, 0, nullptr, sbuf, nullptr, 256, 256, HID);
    sink_init_k<<<dim3(64 * 256), thr, 0, stream>>>(sbuf, n1, n2, tau_p);
    for (int it = 0; it < 10; ++it) {
        if ((it & 1) == 0) sink_row_k<<<dim3(256, 64), thr, 0, stream>>>(sbuf, n1);
        else               sink_col_k<<<dim3(256, 64), thr, 0, stream>>>(sbuf, n2);
    }
    sink_final_k<<<dim3(64 * 256), thr, 0, stream>>>(sbuf, nullptr, s_bf, s_tbf, n1, n2);

    // ---- cross-graph update ----
    G<GF_OUTB>(stream, gBat, s_bf, 65536, 256, emb2t, (long long)HID * 256, 256,
        nullptr, nullptr, 0, tmp1, nullptr, nullptr, HID, HID, 256);
    G<GF_OUTB>(stream, gBat, s_tbf, 65536, 256, emb1t, (long long)HID * 256, 256,
        nullptr, nullptr, 0, tmp2, nullptr, nullptr, HID, HID, 256);
    // new_emb1 = emb1 @ Wtop + tmp1 @ Wbot + b
    G<GF_OUTF>(stream, gFlat, emb1, 0, HID, w_crT, 0, HID,
        nullptr, nullptr, 0, nullptr, P, nullptr, HID, HID, HID);
    G<GF_BIAS | GF_ADDF | GF_OUTB>(stream, gFlat, tmp1, 0, HID, w_crB, 0, HID,
        cross_b, P, HID, ne1, nullptr, nullptr, HID, HID, HID);
    G<GF_OUTF>(stream, gFlat, emb2, 0, HID, w_crT, 0, HID,
        nullptr, nullptr, 0, nullptr, P, nullptr, HID, HID, HID);
    G<GF_BIAS | GF_ADDF | GF_OUTB>(stream, gFlat, tmp2, 0, HID, w_crB, 0, HID,
        cross_b, P, HID, ne2, nullptr, nullptr, HID, HID, HID);

    // ---- layer 1: Gconv (reuse ax/ux/emb buffers) ----
    G<GF_RELU | GF_BIAS | GF_OUTT>(stream, gFlat, ne1, 0, HID, w_g1a, 0, HID,
        g1_ab, nullptr, 0, nullptr, nullptr, ax1t, HID, HID, HID);
    G<GF_RELU | GF_BIAS | GF_OUTB>(stream, gFlat, ne1, 0, HID, w_g1u, 0, HID,
        g1_ub, nullptr, 0, ux1, nullptr, nullptr, HID, HID, HID);
    G<GF_RELU | GF_BIAS | GF_OUTT>(stream, gFlat, ne2, 0, HID, w_g1a, 0, HID,
        g1_ab, nullptr, 0, nullptr, nullptr, ax2t, HID, HID, HID);
    G<GF_RELU | GF_BIAS | GF_OUTB>(stream, gFlat, ne2, 0, HID, w_g1u, 0, HID,
        g1_ub, nullptr, 0, ux2, nullptr, nullptr, HID, HID, HID);
    G<GF_ADDB | GF_OUTB>(stream, gBat, An1, 65536, 256, ax1t, (long long)HID * 256, 256,
        nullptr, ux1, HID, emb1, nullptr, nullptr, HID, HID, 256);
    G<GF_ADDB | GF_OUTB>(stream, gBat, An2, 65536, 256, ax2t, (long long)HID * 256, 256,
        nullptr, ux2, HID, emb2, nullptr, nullptr, HID, HID, 256);

    // ---- affinity 1 + final Sinkhorn ----
    G<GF_OUTB>(stream, gFlat, emb1, 0, HID, w_af1, 0, HID,
        nullptr, nullptr, 0, Tbuf, nullptr, nullptr, HID, HID, HID);
    G<GF_OUTF>(stream, gS, Tbuf, 256LL * HID, HID, emb2, 256LL * HID, HID,
        nullptr, nullptr, 0, nullptr, sbuf, nullptr, 256, 256, HID);
    sink_init_k<<<dim3(64 * 256), thr, 0, stream>>>(sbuf, n1, n2, tau_p);
    for (int it = 0; it < 10; ++it) {
        if ((it & 1) == 0) sink_row_k<<<dim3(256, 64), thr, 0, stream>>>(sbuf, n1);
        else               sink_col_k<<<dim3(256, 64), thr, 0, stream>>>(sbuf, n2);
    }
    sink_final_k<<<dim3(64 * 256), thr, 0, stream>>>(sbuf, out, nullptr, nullptr, n1, n2);
}
